// DualContrastiveModel_84241488544072
// MI455X (gfx1250) — matrix operand order repaired and checked by reading, not compiled, not run on hardware
//
#include <hip/hip_runtime.h>

// Problem sizes (fixed by the reference).
#define NB 512   // batches
#define NN 256   // nodes
#define ND 128   // feature dim

typedef __attribute__((ext_vector_type(16))) __bf16 v16bf;
typedef __attribute__((ext_vector_type(4)))  __bf16 bf16x4;
typedef __attribute__((ext_vector_type(8)))  float  v8f;
typedef __attribute__((ext_vector_type(4)))  float  f32x4;
typedef __attribute__((ext_vector_type(4)))  unsigned int u32x4;

union Frag {
  struct { u32x4 lo, hi; } u;
  v16bf f;
};

// A-operand fragment (16x32 bf16): lane holds M=lane%16; k chunks at
// [kbase, kbase+8) and [kbase+16, kbase+24) with kbase = (lane<16 ? 0 : 8).
// -> two non-contiguous 16B loads at p and p+16 elements.
__device__ __forceinline__ v16bf load_frag_split(const __bf16* p) {
  Frag x;
  x.u.lo = *(const u32x4*)(p);
  x.u.hi = *(const u32x4*)(p + 16);
  return x.f;
}

// A/B fragment in this code's contiguous k order (lane holds row/col lane%16); k values
// 16 consecutive starting at kbase = (lane<16 ? 0 : 16).
// -> one contiguous 32B load.
__device__ __forceinline__ v16bf load_frag_contig(const __bf16* p) {
  Frag x;
  x.u.lo = *(const u32x4*)(p);
  x.u.hi = *(const u32x4*)(p + 8);
  return x.f;
}

__device__ __forceinline__ v8f wmma_bf16(v16bf a, v16bf b, v8f c) {
  return __builtin_amdgcn_wmma_f32_16x16x32_bf16(
      /*neg_a=*/false, a, /*neg_b=*/false, b,
      /*c_mod=*/(short)0, c, /*reuse_a=*/false, /*reuse_b=*/false);
}

// element offset of s_hw[rel] from s_hw[0] at fixed row
__device__ __forceinline__ constexpr int rel_off(int r) { return r * 32 * ND; }

// ---------------------------------------------------------------------------
// Pre-pass: h (f32, [B,N,D]) -> hb (bf16, [B,N,D]) and hbt (bf16, [B,D,N]).
// One-time cost ~134MB of traffic; lets the main kernel read WMMA B-operand
// fragments straight from (L2-resident) global memory in both orientations.
// ---------------------------------------------------------------------------
__global__ __launch_bounds__(128)
void conv_bf16_kernel(const float* __restrict__ h,
                      __bf16* __restrict__ hb,
                      __bf16* __restrict__ hbt) {
  __shared__ __align__(16) __bf16 tile[32][ND];   // 8 KB
  const int wg  = blockIdx.x;        // NB * (NN/32)
  const int b   = wg >> 3;
  const int n0  = (wg & 7) << 5;
  const int tid = threadIdx.x;
  const int dq  = (tid & 31) << 2;
  const int r0  = tid >> 5;

#pragma unroll
  for (int k = 0; k < 8; ++k) {
    const int row = r0 + 4 * k;
    const f32x4 hv = *(const f32x4*)(h + ((size_t)b * NN + n0 + row) * ND + dq);
    bf16x4 w;
    w[0] = (__bf16)hv[0]; w[1] = (__bf16)hv[1];
    w[2] = (__bf16)hv[2]; w[3] = (__bf16)hv[3];
    *(bf16x4*)&tile[row][dq] = w;
    *(bf16x4*)(hb + ((size_t)b * NN + n0 + row) * ND + dq) = w;
  }
  __syncthreads();

  const int d = tid;                 // 0..127
  __bf16* dst = hbt + ((size_t)b * ND + d) * NN + n0;
#pragma unroll
  for (int g = 0; g < 8; ++g) {
    bf16x4 w;
    w[0] = tile[g * 4 + 0][d];
    w[1] = tile[g * 4 + 1][d];
    w[2] = tile[g * 4 + 2][d];
    w[3] = tile[g * 4 + 3][d];
    *(bf16x4*)(dst + g * 4) = w;     // coalesced 64B per wave-quad
  }
}

// ---------------------------------------------------------------------------
// Fast path: B-operand fragments read directly from hb / hbt (global, L2).
// LDS only holds hw_r (A operand) and the per-wave prob transpose buffer.
// No __syncthreads in the j-loop.
// ---------------------------------------------------------------------------
__global__ __launch_bounds__(64)
void gat_fast_kernel(const float* __restrict__ h,
                     const int*   __restrict__ adj,
                     const float* __restrict__ a0,
                     const float* __restrict__ a1,
                     const float* __restrict__ a2,
                     const float* __restrict__ a3,
                     const __bf16* __restrict__ hb,
                     const __bf16* __restrict__ hbt,
                     float* __restrict__ out) {
  __shared__ __align__(16) __bf16 s_hw[4][32][ND];   // 32 KB
  __shared__ __align__(16) __bf16 s_p[2][16][32];    //  2 KB

  const int wg   = blockIdx.x;
  const int b    = wg >> 3;          // 8 i-tiles of 32 rows per batch
  const int i0   = (wg & 7) << 5;
  const int tid  = threadIdx.x;
  const int wave = tid >> 5;
  const int lane = tid & 31;
  const int half = lane >> 4;        // 0: lanes 0-15, 1: lanes 16-31
  const int ln   = lane & 15;

  // ---- precompute hw_r = h[i-tile] * a_r into LDS (bf16), static offsets ---
  {
    const int dq = lane << 2;        // 0..124, constant per thread
    const float* hrow = h + ((size_t)b * NN + i0 + wave) * ND + dq;
#pragma unroll
    for (int rel = 0; rel < 4; ++rel) {
      const float* ap = (rel == 0) ? a0 : (rel == 1) ? a1 : (rel == 2) ? a2 : a3;
      const f32x4 av = *(const f32x4*)(ap + dq);   // one vector load per rel
#pragma unroll
      for (int k = 0; k < 16; ++k) {
        const int row = 2 * k + wave;
        const f32x4 hv = *(const f32x4*)(hrow + (size_t)(2 * k) * ND);
        bf16x4 w;
        w[0] = (__bf16)(hv[0] * av[0]);
        w[1] = (__bf16)(hv[1] * av[1]);
        w[2] = (__bf16)(hv[2] * av[2]);
        w[3] = (__bf16)(hv[3] * av[3]);
        *(bf16x4*)&s_hw[rel][row][dq] = w;         // single ds_store_b64
      }
    }
  }
  __syncthreads();

  // Softmax with fixed reference max 0 (scores bounded; clamped at 60):
  // p = exp(s), masked -> 0; denominator = per-lane partial, reduced once.
  v8f   oacc[8];
  float lrow[8];
#pragma unroll
  for (int q = 0; q < 8; ++q) lrow[q] = 0.f;
#pragma unroll
  for (int nt = 0; nt < 8; ++nt)
#pragma unroll
    for (int q = 0; q < 8; ++q) oacc[nt][q] = 0.f;

  // per-thread bases (j0 added per iteration)
  const int*    adjbase = adj + ((size_t)b * NN + i0 + wave * 16 + 8 * half) * NN + ln;
  const __bf16* hbbase  = hb  + ((size_t)b * NN + ln) * ND + half * 16;
  const __bf16* hbtbase = hbt + ((size_t)b * ND + ln) * NN + half * 16;
  const __bf16* hwbase  = &s_hw[0][wave * 16 + ln][half * 16];

  for (int jt = 0; jt < 8; ++jt) {
    const int j0 = jt << 5;

    // adj gather (C layout: (M,N) <-> (reg,lane)); immediate offsets
    const int* adjp = adjbase + j0;
    int adjv[2][8];
#pragma unroll
    for (int t = 0; t < 2; ++t)
#pragma unroll
      for (int q = 0; q < 8; ++q)
        adjv[t][q] = adjp[q * NN + t * 16];

    // ---- score GEMMs: e_r[16i x 16j] over D=128 (4 chunks of K=32) --------
    const __bf16* pb = hbbase + (size_t)j0 * ND;   // row j0+ln, this lane-half
    v8f e[4][2];
#pragma unroll
    for (int r = 0; r < 4; ++r)
#pragma unroll
      for (int t = 0; t < 2; ++t)
#pragma unroll
        for (int q = 0; q < 8; ++q) e[r][t][q] = 0.f;

#pragma unroll
    for (int c = 0; c < 4; ++c) {
      const v16bf bf0 = load_frag_contig(pb + c * 32);             // j tile 0
      const v16bf bf1 = load_frag_contig(pb + 16 * ND + c * 32);   // j tile 1
#pragma unroll
      for (int r = 0; r < 4; ++r) {
        const v16bf af = load_frag_contig(hwbase + rel_off(r) + c * 32);
        e[r][0] = wmma_bf16(af, bf0, e[r][0]);
        e[r][1] = wmma_bf16(af, bf1, e[r][1]);
      }
    }

    // ---- relation select + leaky-relu + mask + exp ------------------------
    float pv[2][8];
#pragma unroll
    for (int t = 0; t < 2; ++t)
#pragma unroll
      for (int q = 0; q < 8; ++q) {
        const int a = adjv[t][q];
        float x = (a == 1) ? e[0][t][q]
                : (a == 2) ? e[1][t][q]
                : (a == 3) ? e[2][t][q]
                           : e[3][t][q];
        x = fmaxf(x, 0.2f * x);                     // leaky-relu
        x = fminf(x, 60.f);                         // overflow insurance
        const float pe = __expf(x);
        pv[t][q] = ((unsigned)(a - 1) < 4u) ? pe : 0.f;
      }
#pragma unroll
    for (int q = 0; q < 8; ++q) lrow[q] += pv[0][q] + pv[1][q];

    // ---- C-layout -> A-layout transpose of probs via per-wave LDS ---------
#pragma unroll
    for (int t = 0; t < 2; ++t)
#pragma unroll
      for (int q = 0; q < 8; ++q)
        s_p[wave][q + 8 * half][t * 16 + ln] = (__bf16)pv[t][q];
    __asm__ volatile("s_wait_dscnt 0" ::: "memory");  // same-wave DS in order

    const v16bf pA = load_frag_contig(&s_p[wave][ln][half * 16]);

    // ---- output GEMM: oacc[16i x 128d] += p(16x32) x h_j(32x128) ----------
    const __bf16* pbt = hbtbase + j0;               // row d=ln, cols j0+...
#pragma unroll
    for (int nt = 0; nt < 8; ++nt) {
      const v16bf bB = load_frag_contig(pbt + nt * 16 * NN);
      oacc[nt] = wmma_bf16(pA, bB, oacc[nt]);
    }
  }

  // ---- finalize: reduce denominators (rows live in 16-lane halves) --------
#pragma unroll
  for (int q = 0; q < 8; ++q) {
    float l = lrow[q];
    l += __shfl_xor(l, 1);
    l += __shfl_xor(l, 2);
    l += __shfl_xor(l, 4);
    l += __shfl_xor(l, 8);
    const float inv = (l > 0.f) ? (1.0f / l) : 0.f;
    const int gi = i0 + wave * 16 + q + 8 * half;
    float* orow = out + ((size_t)b * NN + gi) * ND;
#pragma unroll
    for (int nt = 0; nt < 8; ++nt)
      orow[nt * 16 + ln] = oacc[nt][q] * inv;
  }
}

// ---------------------------------------------------------------------------
// Fallback (no workspace): self-staging kernel, correct standalone.
// ---------------------------------------------------------------------------
__global__ __launch_bounds__(64)
void gat_fallback_kernel(const float* __restrict__ h,
                         const int*   __restrict__ adj,
                         const float* __restrict__ a0,
                         const float* __restrict__ a1,
                         const float* __restrict__ a2,
                         const float* __restrict__ a3,
                         float* __restrict__ out) {
  __shared__ __align__(16) __bf16 s_hw[4][32][ND];
  __shared__ __align__(16) __bf16 s_hj[32][ND];
  __shared__ __align__(16) __bf16 s_hjt[ND][32];
  __shared__ __align__(16) __bf16 s_p[2][16][32];

  const int wg   = blockIdx.x;
  const int b    = wg >> 3;
  const int i0   = (wg & 7) << 5;
  const int tid  = threadIdx.x;
  const int wave = tid >> 5;
  const int lane = tid & 31;
  const int half = lane >> 4;
  const int ln   = lane & 15;

  for (int v = tid; v < 4 * 32 * (ND / 4); v += 64) {
    const int rel = v >> 10;
    const int row = (v >> 5) & 31;
    const int dq  = (v & 31) << 2;
    const float* ap = (rel == 0) ? a0 : (rel == 1) ? a1 : (rel == 2) ? a2 : a3;
    const f32x4 hv = *(const f32x4*)(h + ((size_t)b * NN + i0 + row) * ND + dq);
    const f32x4 av = *(const f32x4*)(ap + dq);
    bf16x4 w;
    w[0] = (__bf16)(hv[0] * av[0]);
    w[1] = (__bf16)(hv[1] * av[1]);
    w[2] = (__bf16)(hv[2] * av[2]);
    w[3] = (__bf16)(hv[3] * av[3]);
    *(bf16x4*)&s_hw[rel][row][dq] = w;
  }

  v8f   oacc[8];
  float lrow[8];
#pragma unroll
  for (int q = 0; q < 8; ++q) lrow[q] = 0.f;
#pragma unroll
  for (int nt = 0; nt < 8; ++nt)
#pragma unroll
    for (int q = 0; q < 8; ++q) oacc[nt][q] = 0.f;

  for (int jt = 0; jt < 8; ++jt) {
    const int j0 = jt << 5;
    __syncthreads();

    for (int v = tid; v < 32 * (ND / 4); v += 64) {
      const int row = v >> 5;
      const int dq  = (v & 31) << 2;
      const f32x4 hv = *(const f32x4*)(h + ((size_t)b * NN + j0 + row) * ND + dq);
      bf16x4 w;
      w[0] = (__bf16)hv[0]; w[1] = (__bf16)hv[1];
      w[2] = (__bf16)hv[2]; w[3] = (__bf16)hv[3];
      *(bf16x4*)&s_hj[row][dq] = w;
      s_hjt[dq + 0][row] = w[0];
      s_hjt[dq + 1][row] = w[1];
      s_hjt[dq + 2][row] = w[2];
      s_hjt[dq + 3][row] = w[3];
    }

    const int* adjrow =
        adj + ((size_t)b * NN + i0 + wave * 16 + 8 * half) * NN + j0 + ln;
    int adjv[2][8];
#pragma unroll
    for (int t = 0; t < 2; ++t)
#pragma unroll
      for (int q = 0; q < 8; ++q)
        adjv[t][q] = adjrow[q * NN + t * 16];
    __syncthreads();

    v8f e[4][2];
#pragma unroll
    for (int r = 0; r < 4; ++r)
#pragma unroll
      for (int t = 0; t < 2; ++t)
#pragma unroll
        for (int q = 0; q < 8; ++q) e[r][t][q] = 0.f;

#pragma unroll
    for (int c = 0; c < 4; ++c) {
      const v16bf bf0 = load_frag_contig(&s_hj[ln][c * 32 + half * 16]);
      const v16bf bf1 = load_frag_contig(&s_hj[16 + ln][c * 32 + half * 16]);
#pragma unroll
      for (int r = 0; r < 4; ++r) {
        const v16bf af =
            load_frag_contig(&s_hw[r][wave * 16 + ln][c * 32 + half * 16]);
        e[r][0] = wmma_bf16(af, bf0, e[r][0]);
        e[r][1] = wmma_bf16(af, bf1, e[r][1]);
      }
    }

    float pv[2][8];
#pragma unroll
    for (int t = 0; t < 2; ++t)
#pragma unroll
      for (int q = 0; q < 8; ++q) {
        const int a = adjv[t][q];
        float x = (a == 1) ? e[0][t][q]
                : (a == 2) ? e[1][t][q]
                : (a == 3) ? e[2][t][q]
                           : e[3][t][q];
        x = fmaxf(x, 0.2f * x);
        x = fminf(x, 60.f);
        const float pe = __expf(x);
        pv[t][q] = ((unsigned)(a - 1) < 4u) ? pe : 0.f;
      }
#pragma unroll
    for (int q = 0; q < 8; ++q) lrow[q] += pv[0][q] + pv[1][q];

#pragma unroll
    for (int t = 0; t < 2; ++t)
#pragma unroll
      for (int q = 0; q < 8; ++q)
        s_p[wave][q + 8 * half][t * 16 + ln] = (__bf16)pv[t][q];
    __asm__ volatile("s_wait_dscnt 0" ::: "memory");

    const v16bf pA = load_frag_contig(&s_p[wave][ln][half * 16]);

#pragma unroll
    for (int nt = 0; nt < 8; ++nt) {
      const v16bf bB = load_frag_contig(&s_hjt[nt * 16 + ln][half * 16]);
      oacc[nt] = wmma_bf16(pA, bB, oacc[nt]);
    }
  }

#pragma unroll
  for (int q = 0; q < 8; ++q) {
    float l = lrow[q];
    l += __shfl_xor(l, 1);
    l += __shfl_xor(l, 2);
    l += __shfl_xor(l, 4);
    l += __shfl_xor(l, 8);
    const float inv = (l > 0.f) ? (1.0f / l) : 0.f;
    const int gi = i0 + wave * 16 + q + 8 * half;
    float* orow = out + ((size_t)b * NN + gi) * ND;
#pragma unroll
    for (int nt = 0; nt < 8; ++nt)
      orow[nt * 16 + ln] = oacc[nt][q] * inv;
  }
}

extern "C" void kernel_launch(void* const* d_in, const int* in_sizes, int n_in,
                              void* d_out, int out_size, void* d_ws, size_t ws_size,
                              hipStream_t stream) {
  const float* h   = (const float*)d_in[0];
  const int*   adj = (const int*)d_in[1];
  const float* a0  = (const float*)d_in[2];
  const float* a1  = (const float*)d_in[3];
  const float* a2  = (const float*)d_in[4];
  const float* a3  = (const float*)d_in[5];
  float* out = (float*)d_out;

  const size_t elems = (size_t)NB * NN * ND;
  const size_t need  = 2 * elems * sizeof(__bf16);   // hb + hbt = 64 MiB

  if (ws_size >= need) {
    __bf16* hb  = (__bf16*)d_ws;
    __bf16* hbt = hb + elems;
    conv_bf16_kernel<<<dim3(NB * (NN / 32)), dim3(128), 0, stream>>>(h, hb, hbt);
    gat_fast_kernel<<<dim3(NB * (NN / 32)), dim3(64), 0, stream>>>(
        h, adj, a0, a1, a2, a3, hb, hbt, out);
  } else {
    gat_fallback_kernel<<<dim3(NB * (NN / 32)), dim3(64), 0, stream>>>(
        h, adj, a0, a1, a2, a3, out);
  }
}
